// CausualSelfAttention_43224550867757
// MI455X (gfx1250) — matrix operand order repaired and checked by reading, not compiled, not run on hardware
//
#include <hip/hip_runtime.h>

#define B_   2
#define T_   2048
#define C_   1024
#define C3_  (3 * C_)
#define NH_  16
#define HS_  64

typedef __attribute__((ext_vector_type(16))) __bf16        v16bf;
typedef __attribute__((ext_vector_type(8)))  float         v8f;
typedef __attribute__((ext_vector_type(4)))  unsigned int  u32x4;

#if defined(__has_builtin)
#if __has_builtin(__builtin_amdgcn_global_load_async_to_lds_b128)
#define HAVE_ASYNC_LDS 1
#endif
#endif

#if HAVE_ASYNC_LDS
typedef __attribute__((ext_vector_type(4))) int            ai4;
typedef __attribute__((address_space(1))) ai4*             g_ai4p;   // global v4i*
typedef __attribute__((address_space(3))) ai4*             l_ai4p;   // LDS v4i*
#endif

union Frag16 {
    v16bf  v;
    u32x4  q[2];
    __bf16 e[16];
};

static __device__ __forceinline__ __bf16 f2bf(float f) { return (__bf16)f; }

// ---------------------------------------------------------------------------
// Kernel 1: qkv = x @ W_attn + b_attn, emitted as bf16 Q/K/V in [B,NH,T,HS].
// Block = 64x64 output tile, 128 threads (4 waves), K-step = 32.
// ---------------------------------------------------------------------------
__global__ __launch_bounds__(128)
void qkv_wmma_kernel(const float* __restrict__ x, const float* __restrict__ W,
                     const float* __restrict__ bias,
                     __bf16* __restrict__ Qb, __bf16* __restrict__ Kb,
                     __bf16* __restrict__ Vb)
{
    __shared__ __bf16 aLds[64][32];   // A tile: [m][k]
    __shared__ __bf16 bLds[64][32];   // B tile transposed: [n][k]

    const int tid  = threadIdx.x;
    const int lane = tid & 31;
    const int wave = tid >> 5;
    const int ln   = lane & 15;
    const int g    = lane >> 4;

    const int m0 = blockIdx.x * 64;   // row tile over B*T = 4096
    const int n0 = blockIdx.y * 64;   // col tile over 3C = 3072

    v8f acc[4] = {};

    for (int k0 = 0; k0 < C_; k0 += 32) {
        __syncthreads();
        // stage A: 64 rows x 32 k, f32 -> bf16; 16 f32 per thread
        {
            const int row  = tid >> 1;
            const int half = tid & 1;
            const float* src = x + (size_t)(m0 + row) * C_ + k0 + half * 16;
            #pragma unroll
            for (int i = 0; i < 16; ++i) aLds[row][half * 16 + i] = f2bf(src[i]);
        }
        // stage B transposed: W[k0+kr][n0 + qv*16 + i] -> bLds[n][k]
        {
            const int kr = tid >> 2;
            const int qv = tid & 3;
            const float* src = W + (size_t)(k0 + kr) * C3_ + n0 + qv * 16;
            #pragma unroll
            for (int i = 0; i < 16; ++i) bLds[qv * 16 + i][kr] = f2bf(src[i]);
        }
        __syncthreads();

        // A fragment (16x32), rows [wave*16 .. wave*16+15]
        Frag16 af;
        {
            const int row = wave * 16 + ln;
            const int kk  = g * 8;
            af.q[0] = *(const u32x4*)&aLds[row][kk];
            af.q[1] = *(const u32x4*)&aLds[row][kk + 16];
        }
        #pragma unroll
        for (int ns = 0; ns < 4; ++ns) {
            Frag16 bf;
            const int col = ns * 16 + ln;
            const int kk  = g * 8;
            bf.q[0] = *(const u32x4*)&bLds[col][kk];
            bf.q[1] = *(const u32x4*)&bLds[col][kk + 16];
            acc[ns] = __builtin_amdgcn_wmma_f32_16x16x32_bf16(
                false, af.v, false, bf.v, (short)0, acc[ns], false, false);
        }
    }

    // epilogue: + bias, scatter bf16 into Q/K/V [B,NH,T,HS]
    #pragma unroll
    for (int ns = 0; ns < 4; ++ns) {
        const int n   = n0 + ns * 16 + ln;
        const float bv = bias[n];
        const int sec = n / C_;
        const int rem = n % C_;
        const int h   = rem / HS_;
        const int d   = rem % HS_;
        __bf16* dst = (sec == 0) ? Qb : (sec == 1) ? Kb : Vb;
        #pragma unroll
        for (int r = 0; r < 8; ++r) {
            const int m  = m0 + wave * 16 + g * 8 + r;   // global row in B*T
            const int bb = m >> 11;                      // / T_
            const int t  = m & (T_ - 1);
            dst[(((size_t)bb * NH_ + h) * T_ + t) * HS_ + d] = f2bf(acc[ns][r] + bv);
        }
    }
}

// ---------------------------------------------------------------------------
// Kernel 2: causal flash attention over bf16 Q/K/V, f32 output [B,T,C].
// Block = one (b,h) x 64 query rows; wave owns 16 rows; 32-key tiles.
// K/V tiles staged with CDNA5 async global->LDS copies when available.
// ---------------------------------------------------------------------------
__global__ __launch_bounds__(128)
void attn_wmma_kernel(const __bf16* __restrict__ Qb, const __bf16* __restrict__ Kb,
                      const __bf16* __restrict__ Vb, float* __restrict__ out)
{
    __shared__ __bf16 kTile[32][64];
    __shared__ __bf16 vTile[32][64];
    __shared__ __bf16 pLds[4][16][32];

    const int tid  = threadIdx.x;
    const int lane = tid & 31;
    const int wave = tid >> 5;
    const int ln   = lane & 15;
    const int g    = lane >> 4;

    const int q0      = blockIdx.x * 64;       // query tile
    const int bh      = blockIdx.y;            // b*NH + h
    const size_t base = (size_t)bh * T_ * HS_;

    // load this wave's Q fragments (16x64), scale by HS^-0.5 = 0.125 (exact)
    Frag16 qf[2];
    {
        const int qrow = q0 + wave * 16 + ln;
        const __bf16* src = Qb + base + (size_t)qrow * HS_;
        #pragma unroll
        for (int h2 = 0; h2 < 2; ++h2) {
            #pragma unroll
            for (int i = 0; i < 8; ++i) {
                qf[h2].e[i]     = f2bf((float)src[h2 * 32 + g * 8 + i] * 0.125f);
                qf[h2].e[8 + i] = f2bf((float)src[h2 * 32 + 16 + g * 8 + i] * 0.125f);
            }
        }
    }

    v8f o[4] = {};
    float mrun[8], lrun[8];
    #pragma unroll
    for (int r = 0; r < 8; ++r) { mrun[r] = -1e30f; lrun[r] = 0.0f; }

    const int ktmax = (q0 + 64) / 32;

    for (int kt = 0; kt < ktmax; ++kt) {
        __syncthreads();
        // stage 32 keys x 64 d of K and V (bf16): pure copy, no conversion
        {
            const int key = tid >> 2;
            const int qu  = tid & 3;
            const size_t off = base + (size_t)(kt * 32 + key) * HS_ + qu * 16;
#if HAVE_ASYNC_LDS
            __builtin_amdgcn_global_load_async_to_lds_b128(
                (g_ai4p)(Kb + off),     (l_ai4p)&kTile[key][qu * 16],     0, 0);
            __builtin_amdgcn_global_load_async_to_lds_b128(
                (g_ai4p)(Kb + off + 8), (l_ai4p)&kTile[key][qu * 16 + 8], 0, 0);
            __builtin_amdgcn_global_load_async_to_lds_b128(
                (g_ai4p)(Vb + off),     (l_ai4p)&vTile[key][qu * 16],     0, 0);
            __builtin_amdgcn_global_load_async_to_lds_b128(
                (g_ai4p)(Vb + off + 8), (l_ai4p)&vTile[key][qu * 16 + 8], 0, 0);
            asm volatile("s_wait_asynccnt 0x0" ::: "memory");
#else
            const u32x4* ks = (const u32x4*)(Kb + off);
            const u32x4* vs = (const u32x4*)(Vb + off);
            *(u32x4*)&kTile[key][qu * 16]     = ks[0];
            *(u32x4*)&kTile[key][qu * 16 + 8] = ks[1];
            *(u32x4*)&vTile[key][qu * 16]     = vs[0];
            *(u32x4*)&vTile[key][qu * 16 + 8] = vs[1];
#endif
        }
        __syncthreads();

        // S = Q K^T for the 32-key tile: 2 key-halves x 2 d-halves
        v8f s0 = {}, s1 = {};
        #pragma unroll
        for (int dh = 0; dh < 2; ++dh) {
            Frag16 b0, b1;
            const int dd = dh * 32 + g * 8;
            b0.q[0] = *(const u32x4*)&kTile[ln][dd];
            b0.q[1] = *(const u32x4*)&kTile[ln][dd + 16];
            b1.q[0] = *(const u32x4*)&kTile[16 + ln][dd];
            b1.q[1] = *(const u32x4*)&kTile[16 + ln][dd + 16];
            s0 = __builtin_amdgcn_wmma_f32_16x16x32_bf16(
                false, qf[dh].v, false, b0.v, (short)0, s0, false, false);
            s1 = __builtin_amdgcn_wmma_f32_16x16x32_bf16(
                false, qf[dh].v, false, b1.v, (short)0, s1, false, false);
        }

        // causal mask + online softmax; row of element r is qrow_base + r
        const int qrow_base = q0 + wave * 16 + g * 8;
        const int k0g = kt * 32 + ln;
        const int k1g = kt * 32 + 16 + ln;
        float cf[8];
        #pragma unroll
        for (int r = 0; r < 8; ++r) {
            const int qr = qrow_base + r;
            float v0 = (k0g <= qr) ? s0[r] : -1e30f;
            float v1 = (k1g <= qr) ? s1[r] : -1e30f;
            float mx = fmaxf(v0, v1);
            #pragma unroll
            for (int off = 1; off < 16; off <<= 1)
                mx = fmaxf(mx, __shfl_xor(mx, off, 16));
            const float mnew = fmaxf(mrun[r], mx);
            cf[r] = __expf(mrun[r] - mnew);
            const float p0 = __expf(v0 - mnew);
            const float p1 = __expf(v1 - mnew);
            float sum = p0 + p1;
            #pragma unroll
            for (int off = 1; off < 16; off <<= 1)
                sum += __shfl_xor(sum, off, 16);
            lrun[r] = lrun[r] * cf[r] + sum;
            mrun[r] = mnew;
            s0[r] = p0;
            s1[r] = p1;
        }

        // rescale accumulator rows
        #pragma unroll
        for (int d4 = 0; d4 < 4; ++d4)
            #pragma unroll
            for (int r = 0; r < 8; ++r) o[d4][r] *= cf[r];

        // P: C-layout -> bf16 A-layout via per-wave LDS round-trip
        #pragma unroll
        for (int r = 0; r < 8; ++r) {
            pLds[wave][g * 8 + r][ln]      = f2bf(s0[r]);
            pLds[wave][g * 8 + r][16 + ln] = f2bf(s1[r]);
        }
        asm volatile("s_wait_dscnt 0x0" ::: "memory");

        Frag16 pf;
        pf.q[0] = *(const u32x4*)&pLds[wave][ln][g * 8];
        pf.q[1] = *(const u32x4*)&pLds[wave][ln][g * 8 + 16];

        // O += P @ V   (4 d-slabs of 16)
        #pragma unroll
        for (int d4 = 0; d4 < 4; ++d4) {
            Frag16 vf;
            const int dcol = d4 * 16 + ln;
            #pragma unroll
            for (int i = 0; i < 16; ++i) vf.e[i] = vTile[(i & 8) * 2 + g * 8 + (i & 7)][dcol];
            o[d4] = __builtin_amdgcn_wmma_f32_16x16x32_bf16(
                false, pf.v, false, vf.v, (short)0, o[d4], false, false);
        }
    }

    // normalize and write out[b][q][h*HS + d] (f32)
    const int b = bh / NH_;
    const int h = bh % NH_;
    #pragma unroll
    for (int r = 0; r < 8; ++r) {
        const float inv = 1.0f / lrun[r];
        const int qr = q0 + wave * 16 + g * 8 + r;
        float* dst = out + ((size_t)(b * T_ + qr)) * C_ + h * HS_;
        #pragma unroll
        for (int d4 = 0; d4 < 4; ++d4)
            dst[d4 * 16 + ln] = o[d4][r] * inv;
    }
}

// ---------------------------------------------------------------------------
extern "C" void kernel_launch(void* const* d_in, const int* in_sizes, int n_in,
                              void* d_out, int out_size, void* d_ws, size_t ws_size,
                              hipStream_t stream)
{
    (void)in_sizes; (void)n_in; (void)out_size; (void)ws_size;

    const float* x    = (const float*)d_in[0];
    const float* W    = (const float*)d_in[1];
    const float* bias = (const float*)d_in[2];
    float*       out  = (float*)d_out;

    const size_t nQ = (size_t)B_ * NH_ * T_ * HS_;   // 4,194,304 elems
    __bf16* Qb = (__bf16*)d_ws;
    __bf16* Kb = Qb + nQ;
    __bf16* Vb = Kb + nQ;

    dim3 gGemm(B_ * T_ / 64, C3_ / 64);   // 64 x 48 blocks
    qkv_wmma_kernel<<<gGemm, 128, 0, stream>>>(x, W, bias, Qb, Kb, Vb);

    dim3 gAttn(T_ / 64, B_ * NH_);        // 32 x 32 blocks
    attn_wmma_kernel<<<gAttn, 128, 0, stream>>>(Qb, Kb, Vb, out);
}
